// LinearTemporalSelfAttention_80298708566044
// MI455X (gfx1250) — matrix operand order repaired and checked by reading, not compiled, not run on hardware
//
#include <hip/hip_runtime.h>

// ---------------------------------------------------------------------------
// LinearTemporalSelfAttention for MI455X (gfx1250, wave32, WMMA)
// B=32 T=4096 D=512 H=8 dh=64.  ~292 GFLOP, ~1.2 GB HBM traffic -> run all
// big GEMMs on v_wmma_f32_16x16x32_bf16 so the kernel stays HBM-bound.
// ---------------------------------------------------------------------------

typedef __attribute__((ext_vector_type(16))) __bf16 v16bf;
typedef __attribute__((ext_vector_type(8)))  float  v8f;

constexpr int Bn  = 32;
constexpr int Tn  = 4096;
constexpr int Dn  = 512;
constexpr int Hn  = 8;
constexpr int DHn = 64;
constexpr int Mn  = Bn * Tn;          // 131072 rows
constexpr int NSPLIT = 16;            // T-splits for KV-state partials
constexpr float NEGC = -1000000.0f;

// ------------------------------ helpers ------------------------------------

static __device__ inline unsigned short f32_to_bf16(float f) {
  unsigned int u = __float_as_uint(f);
  unsigned int r = u + 0x7FFFu + ((u >> 16) & 1u);   // round-to-nearest-even
  return (unsigned short)(r >> 16);
}
static __device__ inline float bf16_to_f32(unsigned short h) {
  return __uint_as_float(((unsigned int)h) << 16);
}

union BfVec {
  v16bf v;
  uint4 q4[2];
  unsigned short h[16];
};

static __device__ inline v8f wmma_bf16(const BfVec& a, const BfVec& b, v8f c) {
  return __builtin_amdgcn_wmma_f32_16x16x32_bf16(
      /*neg_a=*/false, a.v, /*neg_b=*/false, b.v,
      /*c_mod=*/(short)0, c, /*reuse_a=*/false, /*reuse_b=*/false);
}

static __device__ inline float wave_sum(float v) {
  #pragma unroll
  for (int o = 16; o > 0; o >>= 1) v += __shfl_xor(v, o, 32);
  return v;
}

// --------------------- weight convert + transpose --------------------------
// W[k][n] (row-major, k-major) -> WT[n][k] bf16, so WMMA B-operand lanes
// (which hold B columns) read two runs of 8 contiguous halves.
__global__ void wconv_kernel(const float* __restrict__ W,
                             unsigned short* __restrict__ WT) {
  int tid = blockIdx.x * blockDim.x + threadIdx.x;     // Dn*Dn threads
  int n = tid >> 9, k = tid & 511;
  WT[(size_t)n * Dn + k] = f32_to_bf16(W[(size_t)k * Dn + n]);
}

// ------------------------------ LayerNorm ----------------------------------
// 1 wave per row of 512; 16 contiguous floats per lane; shuffle reductions.
__global__ __launch_bounds__(256) void ln_kernel(
    const float* __restrict__ x, const float* __restrict__ gamma,
    const float* __restrict__ beta, unsigned short* __restrict__ xn) {
  const int wave = threadIdx.x >> 5, lane = threadIdx.x & 31;
  const size_t row = (size_t)blockIdx.x * 8 + wave;
  const float* px = x + row * Dn + lane * 16;
  union { float4 f4[4]; float f[16]; } u;
  #pragma unroll
  for (int i = 0; i < 4; ++i) u.f4[i] = ((const float4*)px)[i];
  float s = 0.f, sq = 0.f;
  #pragma unroll
  for (int e = 0; e < 16; ++e) { s += u.f[e]; sq += u.f[e] * u.f[e]; }
  s = wave_sum(s); sq = wave_sum(sq);
  const float mu = s * (1.0f / Dn);
  const float var = sq * (1.0f / Dn) - mu * mu;
  const float rs = rsqrtf(var + 1e-5f);
  union { uint4 q4[2]; unsigned short h[16]; } o;
  #pragma unroll
  for (int e = 0; e < 16; ++e) {
    const int c = lane * 16 + e;
    o.h[e] = f32_to_bf16((u.f[e] - mu) * rs * gamma[c] + beta[c]);
  }
  uint4* pd = (uint4*)(xn + row * Dn + lane * 16);
  pd[0] = o.q4[0]; pd[1] = o.q4[1];
}

// ------------------------------ big GEMM -----------------------------------
// C[M,512] = A[M,512] * W  (W given as WT[n][k]).  4 waves/block, each wave
// owns a 64x64 tile (4x4 WMMA accumulators), K-loop in steps of 32.
// mode 0: +bias           -> bf16  (Q logits)
// mode 1: +bias+(1-m)*NEG -> bf16  (K logits)
// mode 2: (+bias)*m       -> bf16  (V)
// mode 3: x + . + bias    -> f32   (final output)
__global__ __launch_bounds__(128) void gemm_kernel(
    const unsigned short* __restrict__ A, const unsigned short* __restrict__ BT,
    const float* __restrict__ bias, const float* __restrict__ mask,
    const float* __restrict__ xres, unsigned short* __restrict__ outb,
    float* __restrict__ outf, int mode) {
  const int wave = threadIdx.x >> 5, lane = threadIdx.x & 31;
  const int lr = lane & 15;
  const int kh8  = (lane >> 4) << 3;    // A-operand K-offset pattern
  const int kh16 = (lane >> 4) << 4;    // unused; B loads use kh8 as well
  const int m0 = blockIdx.x * 256 + wave * 64;
  const int n0 = blockIdx.y * 64;

  const v8f vzero = {0.f, 0.f, 0.f, 0.f, 0.f, 0.f, 0.f, 0.f};
  v8f acc[4][4];
  #pragma unroll
  for (int i = 0; i < 4; ++i)
    #pragma unroll
    for (int j = 0; j < 4; ++j) acc[i][j] = vzero;

  for (int k0 = 0; k0 < Dn; k0 += 32) {
    BfVec a[4], b[4];
    #pragma unroll
    for (int i = 0; i < 4; ++i) {
      const unsigned short* pa = A + (size_t)(m0 + i * 16 + lr) * Dn + k0 + kh8;
      a[i].q4[0] = *(const uint4*)pa;
      a[i].q4[1] = *(const uint4*)(pa + 16);
      __builtin_prefetch(pa + 32, 0, 1);          // global_prefetch_b8
      const unsigned short* pb = BT + (size_t)(n0 + i * 16 + lr) * Dn + k0 + kh8;
      b[i].q4[0] = *(const uint4*)pb;
      b[i].q4[1] = *(const uint4*)(pb + 16);
    }
    #pragma unroll
    for (int i = 0; i < 4; ++i)
      #pragma unroll
      for (int j = 0; j < 4; ++j) acc[i][j] = wmma_bf16(a[i], b[j], acc[i][j]);
  }

  const int rhalf = (lane >> 4) << 3;             // C/D layout: lanes>=16 -> M+8
  #pragma unroll
  for (int i = 0; i < 4; ++i) {
    const int rbase = m0 + i * 16 + rhalf;
    #pragma unroll
    for (int j = 0; j < 4; ++j) {
      const int col = n0 + j * 16 + lr;
      const float bc = bias[col];
      #pragma unroll
      for (int e = 0; e < 8; ++e) {
        const int row = rbase + e;
        float v = acc[i][j][e] + bc;
        if (mode == 1)      v += (1.0f - mask[row]) * NEGC;
        else if (mode == 2) v *= mask[row];
        if (mode == 3) outf[(size_t)row * Dn + col] = xres[(size_t)row * Dn + col] + v;
        else           outb[(size_t)row * Dn + col] = f32_to_bf16(v);
      }
    }
  }
}

// -------------------- Q softmax over head-dim (64) -------------------------
__global__ __launch_bounds__(256) void qsoftmax_kernel(unsigned short* __restrict__ q) {
  const size_t idx = (size_t)blockIdx.x * blockDim.x + threadIdx.x; // Mn*Hn
  const size_t base = (idx >> 3) * Dn + (idx & 7) * DHn;
  union { uint4 q4[8]; unsigned short h[64]; } u;
  uint4* p = (uint4*)(q + base);
  #pragma unroll
  for (int i = 0; i < 8; ++i) u.q4[i] = p[i];
  float m = -INFINITY;
  #pragma unroll
  for (int e = 0; e < 64; ++e) m = fmaxf(m, bf16_to_f32(u.h[e]));
  float s = 0.f;
  #pragma unroll
  for (int e = 0; e < 64; ++e) s += __expf(bf16_to_f32(u.h[e]) - m);
  const float inv = 1.0f / s;
  #pragma unroll
  for (int e = 0; e < 64; ++e)
    u.h[e] = f32_to_bf16(__expf(bf16_to_f32(u.h[e]) - m) * inv);
  #pragma unroll
  for (int i = 0; i < 8; ++i) p[i] = u.q4[i];
}

// ------------------- K softmax over time (per column) ----------------------
// One thread per (b, channel): independent online softmax over T, coalesced
// row-wise reads across the 512 threads of the block.  Deterministic.
__global__ __launch_bounds__(512) void ksoftmax_kernel(unsigned short* __restrict__ k) {
  const int b = blockIdx.x, j = threadIdx.x;
  unsigned short* base = k + (size_t)b * Tn * Dn + j;
  float m = -INFINITY, s = 0.f;
  for (int t = 0; t < Tn; ++t) {
    const float v = bf16_to_f32(base[(size_t)t * Dn]);
    const float nm = fmaxf(m, v);
    s = s * __expf(m - nm) + __expf(v - nm);
    m = nm;
  }
  const float inv = 1.0f / s;
  for (int t = 0; t < Tn; ++t) {
    const float v = bf16_to_f32(base[(size_t)t * Dn]);
    base[(size_t)t * Dn] = f32_to_bf16(__expf(v - m) * inv);
  }
}

// ------------------- KV state: attn[b,h] = K^T V (64x64) -------------------
// One wave per (b,h,split); contraction axis is time, so operand elements are
// per-lane strided gathers (stride D).  Partials are private -> deterministic.
__global__ __launch_bounds__(32) void attn_partial_kernel(
    const unsigned short* __restrict__ K16, const unsigned short* __restrict__ V16,
    float* __restrict__ part) {
  const int lane = threadIdx.x & 31;
  const int bh = blockIdx.x, b = bh >> 3, h = bh & 7;
  const int split = blockIdx.y;
  const int lr = lane & 15;
  const int kh8 = (lane >> 4) << 3, kh16 = (lane >> 4) << 4;
  const unsigned short* kp = K16 + (size_t)b * Tn * Dn + h * DHn;
  const unsigned short* vp = V16 + (size_t)b * Tn * Dn + h * DHn;

  const v8f vzero = {0.f, 0.f, 0.f, 0.f, 0.f, 0.f, 0.f, 0.f};
  v8f acc[4][4];
  #pragma unroll
  for (int i = 0; i < 4; ++i)
    #pragma unroll
    for (int j = 0; j < 4; ++j) acc[i][j] = vzero;

  const int tbeg = split * (Tn / NSPLIT);
  for (int t0 = tbeg; t0 < tbeg + (Tn / NSPLIT); t0 += 32) {
    BfVec a[4], bb[4];
    #pragma unroll
    for (int i = 0; i < 4; ++i) {
      const int d = i * 16 + lr;                 // A row (k-channel)
      #pragma unroll
      for (int e = 0; e < 8; ++e) {              // A[d,t] = k[t,d]
        a[i].h[e]     = kp[(size_t)(t0 + kh8 + e) * Dn + d];
        a[i].h[e + 8] = kp[(size_t)(t0 + kh8 + 16 + e) * Dn + d];
      }
      #pragma unroll
      for (int e = 0; e < 16; ++e)               // B[t,l] column l
        bb[i].h[e] = vp[(size_t)(t0 + kh8 + (e < 8 ? e : e + 8)) * Dn + d];
    }
    #pragma unroll
    for (int i = 0; i < 4; ++i)
      #pragma unroll
      for (int j = 0; j < 4; ++j) acc[i][j] = wmma_bf16(a[i], bb[j], acc[i][j]);
  }

  float* out = part + ((size_t)split * (Bn * Hn) + bh) * (DHn * DHn);
  const int rhalf = (lane >> 4) << 3;
  #pragma unroll
  for (int i = 0; i < 4; ++i)
    #pragma unroll
    for (int j = 0; j < 4; ++j)
      #pragma unroll
      for (int e = 0; e < 8; ++e) {
        const int d = i * 16 + rhalf + e, l = j * 16 + lr;
        out[d * DHn + l] = acc[i][j][e];
      }
}

// Reduce NSPLIT partials; store TRANSPOSED bf16 state attnT[bh][l][d] so the
// apply-GEMM's B-operand (lane=column l, K=d) reads contiguous halves.
__global__ __launch_bounds__(256) void attn_reduce_kernel(
    const float* __restrict__ part, unsigned short* __restrict__ attnT) {
  const size_t tid = (size_t)blockIdx.x * blockDim.x + threadIdx.x; // Bn*Hn*4096
  const int bh = (int)(tid >> 12), r = (int)(tid & 4095);
  const int d = r >> 6, l = r & 63;
  float s = 0.f;
  #pragma unroll
  for (int sp = 0; sp < NSPLIT; ++sp)
    s += part[((size_t)sp * (Bn * Hn) + bh) * 4096 + r];
  attnT[(size_t)bh * 4096 + l * DHn + d] = f32_to_bf16(s);
}

// --------------------- y = Q(softmaxed) @ attn[b,h] ------------------------
__global__ __launch_bounds__(128) void apply_kernel(
    const unsigned short* __restrict__ q, const unsigned short* __restrict__ attnT,
    unsigned short* __restrict__ y) {
  const int wave = threadIdx.x >> 5, lane = threadIdx.x & 31;
  const int bh = blockIdx.y, b = bh >> 3, h = bh & 7;
  const int t0 = blockIdx.x * 256 + wave * 64;
  const int lr = lane & 15;
  const int kh8 = (lane >> 4) << 3, kh16 = (lane >> 4) << 4;
  const unsigned short* qb = q + (size_t)b * Tn * Dn + h * DHn;
  const unsigned short* ab = attnT + (size_t)bh * 4096;

  const v8f vzero = {0.f, 0.f, 0.f, 0.f, 0.f, 0.f, 0.f, 0.f};
  v8f acc[4][4];
  #pragma unroll
  for (int i = 0; i < 4; ++i)
    #pragma unroll
    for (int j = 0; j < 4; ++j) acc[i][j] = vzero;

  #pragma unroll
  for (int k0 = 0; k0 < DHn; k0 += 32) {
    BfVec a[4], bb[4];
    #pragma unroll
    for (int i = 0; i < 4; ++i) {
      const unsigned short* pa = qb + (size_t)(t0 + i * 16 + lr) * Dn + k0 + kh8;
      a[i].q4[0] = *(const uint4*)pa;
      a[i].q4[1] = *(const uint4*)(pa + 16);
      const unsigned short* pb = ab + (size_t)(i * 16 + lr) * DHn + k0 + kh8;
      bb[i].q4[0] = *(const uint4*)pb;
      bb[i].q4[1] = *(const uint4*)(pb + 16);
    }
    #pragma unroll
    for (int i = 0; i < 4; ++i)
      #pragma unroll
      for (int j = 0; j < 4; ++j) acc[i][j] = wmma_bf16(a[i], bb[j], acc[i][j]);
  }

  const int rhalf = (lane >> 4) << 3;
  #pragma unroll
  for (int i = 0; i < 4; ++i)
    #pragma unroll
    for (int j = 0; j < 4; ++j)
      #pragma unroll
      for (int e = 0; e < 8; ++e) {
        const int t = t0 + i * 16 + rhalf + e;
        const int l = j * 16 + lr;
        y[((size_t)b * Tn + t) * Dn + h * DHn + l] = f32_to_bf16(acc[i][j][e]);
      }
}

// ------------------------------ launcher -----------------------------------
extern "C" void kernel_launch(void* const* d_in, const int* in_sizes, int n_in,
                              void* d_out, int out_size, void* d_ws, size_t ws_size,
                              hipStream_t stream) {
  (void)in_sizes; (void)n_in; (void)out_size; (void)ws_size;
  const float* x     = (const float*)d_in[0];
  const float* mask  = (const float*)d_in[1];   // [B,T,1] flat -> per-row
  const float* gamma = (const float*)d_in[2];
  const float* beta  = (const float*)d_in[3];
  const float* Wq    = (const float*)d_in[4];
  const float* bq    = (const float*)d_in[5];
  const float* Wk    = (const float*)d_in[6];
  const float* bk    = (const float*)d_in[7];
  const float* Wv    = (const float*)d_in[8];
  const float* bv    = (const float*)d_in[9];
  const float* Wo    = (const float*)d_in[10];
  const float* bo    = (const float*)d_in[11];
  float* out = (float*)d_out;

  // workspace carve (all sizes 256B-aligned by construction)
  char* w = (char*)d_ws;
  auto take = [&](size_t bytes) { void* p = (void*)w; w += bytes; return p; };
  unsigned short* wqt  = (unsigned short*)take((size_t)Dn * Dn * 2);
  unsigned short* wkt  = (unsigned short*)take((size_t)Dn * Dn * 2);
  unsigned short* wvt  = (unsigned short*)take((size_t)Dn * Dn * 2);
  unsigned short* wot  = (unsigned short*)take((size_t)Dn * Dn * 2);
  unsigned short* xn16 = (unsigned short*)take((size_t)Mn * Dn * 2);
  unsigned short* q16  = (unsigned short*)take((size_t)Mn * Dn * 2);
  unsigned short* k16  = (unsigned short*)take((size_t)Mn * Dn * 2);
  unsigned short* v16  = (unsigned short*)take((size_t)Mn * Dn * 2);
  unsigned short* y16  = (unsigned short*)take((size_t)Mn * Dn * 2);
  float* part          = (float*)take((size_t)NSPLIT * Bn * Hn * DHn * DHn * 4);
  unsigned short* at16 = (unsigned short*)take((size_t)Bn * Hn * DHn * DHn * 2);

  // 1. weights -> bf16 transposed
  wconv_kernel<<<dim3((Dn * Dn) / 256), 256, 0, stream>>>(Wq, wqt);
  wconv_kernel<<<dim3((Dn * Dn) / 256), 256, 0, stream>>>(Wk, wkt);
  wconv_kernel<<<dim3((Dn * Dn) / 256), 256, 0, stream>>>(Wv, wvt);
  wconv_kernel<<<dim3((Dn * Dn) / 256), 256, 0, stream>>>(Wo, wot);

  // 2. LayerNorm -> bf16
  ln_kernel<<<dim3(Mn / 8), 256, 0, stream>>>(x, gamma, beta, xn16);

  // 3. Q/K/V projections (WMMA)
  dim3 gg(Mn / 256, Dn / 64);
  gemm_kernel<<<gg, 128, 0, stream>>>(xn16, wqt, bq, nullptr, nullptr, q16, nullptr, 0);
  gemm_kernel<<<gg, 128, 0, stream>>>(xn16, wkt, bk, mask, nullptr, k16, nullptr, 1);
  gemm_kernel<<<gg, 128, 0, stream>>>(xn16, wvt, bv, mask, nullptr, v16, nullptr, 2);

  // 4. softmaxes
  qsoftmax_kernel<<<dim3((Mn * Hn) / 256), 256, 0, stream>>>(q16);
  ksoftmax_kernel<<<dim3(Bn), 512, 0, stream>>>(k16);

  // 5. KV state (WMMA over time) + deterministic reduction
  attn_partial_kernel<<<dim3(Bn * Hn, NSPLIT), 32, 0, stream>>>(k16, v16, part);
  attn_reduce_kernel<<<dim3((Bn * Hn * DHn * DHn) / 256), 256, 0, stream>>>(part, at16);

  // 6. apply state per token (WMMA)
  apply_kernel<<<dim3(Tn / 256, Bn * Hn), 128, 0, stream>>>(q16, at16, y16);

  // 7. output projection + residual (WMMA, f32 out)
  gemm_kernel<<<gg, 128, 0, stream>>>(y16, wot, bo, nullptr, x, nullptr, out, 3);
}
